// LSTM_Trend_2826088481541
// MI455X (gfx1250) — matrix operand order repaired and checked by reading, not compiled, not run on hardware
//
#include <hip/hip_runtime.h>
#include <hip/hip_bf16.h>
#include <cstdint>

// ---------------------------------------------------------------------------
// LSTM (B=256,T=512,F=64,H=512) + FC head, CDNA5 / gfx1250.
//
// Strategy:
//  * Batch-partitioned persistent kernel: 16 blocks x 16 batch rows. Each
//    block computes ALL 2048 gate columns for its rows -> recurrence is
//    block-local, zero global synchronization across 512 timesteps.
//  * bf16 WMMA (v_wmma_f32_16x16x32_bf16) with f32 accumulation for both the
//    fused input GEMM (K=64) and the recurrent GEMM (K=512).
//  * W_hh/W_ih/W_fc pre-packed (prep kernels) into the ISA 16-bit B-fragment
//    lane layout -> B loads are 2x global_load_b128 per fragment, served from
//    the 192MB L2 (packed W_hh = 2MB, resident).
//  * h double-buffered in LDS (2x16x512 bf16 = 32KB); c state in VGPRs.
// ---------------------------------------------------------------------------

typedef __bf16 v16bf __attribute__((ext_vector_type(16)));
typedef float  v8f   __attribute__((ext_vector_type(8)));

#define BT   16     // batch tile (WMMA M)
#define HH   512    // hidden
#define FF   64     // feature
#define TT   512    // timesteps
#define G4   2048   // 4*H
#define PF   1536   // P*F output cols

// workspace offsets (bytes)
#define WS_WHH  0u                       // 2048*512*2  = 2,097,152
#define WS_WIH  2097152u                 // 2048*64*2   =   262,144
#define WS_WFC  2359296u                 // 1536*512*2  = 1,572,864
#define WS_BIAS 3932160u                 // 2048*4      =     8,192

__device__ __forceinline__ v8f zero_v8() {
  v8f z; 
#pragma unroll
  for (int i = 0; i < 8; ++i) z[i] = 0.0f;
  return z;
}

__device__ __forceinline__ v8f wmma_bf16(v16bf a, v16bf b, v8f c) {
  return __builtin_amdgcn_wmma_f32_16x16x32_bf16(false, a, false, b,
                                                 (short)0, c, false, false);
}

union FragU { v16bf v; uint4 q[2]; };

// A-fragment (16x32 bf16) from a row-major bf16 matrix with leading dim ldk.
// ISA layout: lanes 0-15 -> K = kt*32 + {0..7, 16..23}; lanes 16-31 -> +8.
__device__ __forceinline__ v16bf load_a_frag(const __bf16* base, int ldk,
                                             int kt, int m, int hi8) {
  FragU u;
  const __bf16* p = base + m * ldk + kt * 32 + hi8;     // hi8 in {0,8}
  u.q[0] = *(const uint4*)(p);
  u.q[1] = *(const uint4*)(p + 16);
  return u.v;
}

// B-fragment from pre-packed buffer: tile = nt*nKtiles + kt, 32 lanes x 32B.
__device__ __forceinline__ v16bf load_b_frag(const __bf16* packed,
                                             int tile, int lane) {
  FragU u;
  const __bf16* p = packed + ((size_t)tile * 32 + (size_t)lane) * 16;
  u.q[0] = *(const uint4*)(p);
  u.q[1] = *(const uint4*)(p + 8);
  return u.v;
}

__device__ __forceinline__ float sigmoid_f(float v) {
  return 1.0f / (1.0f + __expf(-v));
}
__device__ __forceinline__ float tanh_f(float v) {
  float e = __expf(-2.0f * v);
  return (1.0f - e) / (1.0f + e);
}

// ---------------------------------------------------------------------------
// Prep: pack row-major f32 W[N][K] into bf16 WMMA B-fragments.
// One wave per 32x16 (KxN) tile.  tile = nt*(K/32) + kt.
// ---------------------------------------------------------------------------
__global__ void pack_b_kernel(const float* __restrict__ W,
                              __bf16* __restrict__ P, int N, int K) {
  const int tile = blockIdx.x;
  const int nK   = K >> 5;
  const int nt   = tile / nK;
  const int kt   = tile - nt * nK;
  const int lane = threadIdx.x;            // 0..31
  const int n    = nt * 16 + (lane & 15);
  const int hi8  = (lane & 16) ? 8 : 0;
  const int kb   = kt * 32 + hi8;

  FragU u;
#pragma unroll
  for (int e = 0; e < 8; ++e)
    u.v[e] = (__bf16)W[(size_t)n * K + kb + e];
#pragma unroll
  for (int e = 0; e < 8; ++e)
    u.v[8 + e] = (__bf16)W[(size_t)n * K + kb + 16 + e];

  __bf16* dst = P + ((size_t)tile * 32 + lane) * 16;
  *(uint4*)(dst)      = u.q[0];
  *(uint4*)(dst + 8)  = u.q[1];
}

__global__ void bias_kernel(const float* __restrict__ b_ih,
                            const float* __restrict__ b_hh,
                            float* __restrict__ bias) {
  int i = blockIdx.x * blockDim.x + threadIdx.x;
  if (i < G4) bias[i] = b_ih[i] + b_hh[i];
}

// ---------------------------------------------------------------------------
// Persistent LSTM + FC head.  grid = B/16 blocks, 256 threads (8 waves).
// Wave w owns hidden columns [w*64, w*64+64) -> 4 j-tiles; for each it keeps
// 4 f32 accumulators (i,f,g,o) sharing one A-fragment stream from LDS.
// ---------------------------------------------------------------------------
__global__ __launch_bounds__(256, 1)
void lstm_persistent(const float* __restrict__ x,
                     const float* __restrict__ bias,
                     const float* __restrict__ b_fc,
                     const __bf16* __restrict__ Whh_p,
                     const __bf16* __restrict__ Wih_p,
                     const __bf16* __restrict__ Wfc_p,
                     float* __restrict__ out) {
  __shared__ __align__(16) __bf16 h_lds[2][BT][HH];   // 32 KB
  __shared__ __align__(16) __bf16 x_lds[BT][FF];      //  2 KB

  const int tid  = threadIdx.x;
  const int lane = tid & 31;
  const int wave = tid >> 5;
  const int b0   = blockIdx.x * BT;
  const int m    = lane & 15;
  const int hi8  = (lane & 16) ? 8 : 0;

  // h_{-1} = 0 in buffer 0
  for (int i = tid; i < BT * HH; i += 256)
    ((__bf16*)h_lds[0])[i] = (__bf16)0.0f;

  v8f c[4];
#pragma unroll
  for (int jt = 0; jt < 4; ++jt) c[jt] = zero_v8();

  for (int t = 0; t < TT; ++t) {
    const int rbuf = t & 1;        // holds h_{t-1}
    const int wbuf = rbuf ^ 1;

    // stage x[:, t, :] as bf16 (coalesced, 4 elems/thread)
    for (int i = tid; i < BT * FF; i += 256) {
      const int row = i >> 6, col = i & 63;
      x_lds[row][col] =
          (__bf16)x[((size_t)(b0 + row) * TT + t) * FF + col];
    }
    __syncthreads();

#pragma unroll 1
    for (int jt = 0; jt < 4; ++jt) {
      const int j16 = wave * 4 + jt;            // hidden n-tile 0..31
      v8f ai = zero_v8(), af = zero_v8(), ag = zero_v8(), ao = zero_v8();

      // fused input GEMM: K = 64  (gate q -> packed n-tile q*32 + j16, nK=2)
#pragma unroll
      for (int kt = 0; kt < 2; ++kt) {
        const v16bf a = load_a_frag(&x_lds[0][0], FF, kt, m, hi8);
        ai = wmma_bf16(a, load_b_frag(Wih_p, (0 * 32 + j16) * 2 + kt, lane), ai);
        af = wmma_bf16(a, load_b_frag(Wih_p, (1 * 32 + j16) * 2 + kt, lane), af);
        ag = wmma_bf16(a, load_b_frag(Wih_p, (2 * 32 + j16) * 2 + kt, lane), ag);
        ao = wmma_bf16(a, load_b_frag(Wih_p, (3 * 32 + j16) * 2 + kt, lane), ao);
      }

      // combined bias (b_ih + b_hh), broadcast per output column
      const int jcol = j16 * 16 + m;
      const float bi = bias[0 * HH + jcol];
      const float bf = bias[1 * HH + jcol];
      const float bg = bias[2 * HH + jcol];
      const float bo = bias[3 * HH + jcol];
#pragma unroll
      for (int r = 0; r < 8; ++r) {
        ai[r] += bi; af[r] += bf; ag[r] += bg; ao[r] += bo;
      }

      // recurrent GEMM: K = 512 (nK = 16), A shared across the 4 gates
      const __bf16* hrd = &h_lds[rbuf][0][0];
#pragma unroll 4
      for (int kt = 0; kt < 16; ++kt) {
        const v16bf a = load_a_frag(hrd, HH, kt, m, hi8);
        ai = wmma_bf16(a, load_b_frag(Whh_p, (0 * 32 + j16) * 16 + kt, lane), ai);
        af = wmma_bf16(a, load_b_frag(Whh_p, (1 * 32 + j16) * 16 + kt, lane), af);
        ag = wmma_bf16(a, load_b_frag(Whh_p, (2 * 32 + j16) * 16 + kt, lane), ag);
        ao = wmma_bf16(a, load_b_frag(Whh_p, (3 * 32 + j16) * 16 + kt, lane), ao);
      }

      // activations + cell/hidden update (f32, in registers)
      v8f cc = c[jt];
      v8f hn;
#pragma unroll
      for (int r = 0; r < 8; ++r) {
        const float iv = sigmoid_f(ai[r]);
        const float fv = sigmoid_f(af[r]);
        const float gv = tanh_f(ag[r]);
        const float ov = sigmoid_f(ao[r]);
        const float cn = fv * cc[r] + iv * gv;
        cc[r] = cn;
        hn[r] = ov * tanh_f(cn);
      }
      c[jt] = cc;

      // scatter h_t tile into write buffer (D layout: row = r + hi8)
#pragma unroll
      for (int r = 0; r < 8; ++r)
        h_lds[wbuf][r + hi8][jcol] = (__bf16)hn[r];
    }
    __syncthreads();
  }

  // ---- FC head: out[16, 1536] = h_T @ W_fc^T + b_fc  (h_T in buffer TT&1) --
  const __bf16* hf = &h_lds[TT & 1][0][0];
  for (int nt = wave; nt < PF / 16; nt += 8) {   // 12 tiles per wave, uniform
    v8f acc = zero_v8();
#pragma unroll 4
    for (int kt = 0; kt < 16; ++kt) {
      const v16bf a = load_a_frag(hf, HH, kt, m, hi8);
      acc = wmma_bf16(a, load_b_frag(Wfc_p, nt * 16 + kt, lane), acc);
    }
    const int n  = nt * 16 + m;
    const float bb = b_fc[n];
#pragma unroll
    for (int r = 0; r < 8; ++r)
      out[(size_t)(b0 + r + hi8) * PF + n] = acc[r] + bb;
  }
}

// ---------------------------------------------------------------------------
extern "C" void kernel_launch(void* const* d_in, const int* in_sizes, int n_in,
                              void* d_out, int out_size, void* d_ws, size_t ws_size,
                              hipStream_t stream) {
  (void)in_sizes; (void)n_in; (void)out_size; (void)ws_size;
  const float* x    = (const float*)d_in[0];   // [256,512,64]
  const float* W_ih = (const float*)d_in[1];   // [2048,64]
  const float* W_hh = (const float*)d_in[2];   // [2048,512]
  const float* b_ih = (const float*)d_in[3];   // [2048]
  const float* b_hh = (const float*)d_in[4];   // [2048]
  const float* W_fc = (const float*)d_in[5];   // [1536,512]
  const float* b_fc = (const float*)d_in[6];   // [1536]
  float* out = (float*)d_out;                  // [256,24,64]

  char* ws = (char*)d_ws;
  __bf16* whh_p = (__bf16*)(ws + WS_WHH);
  __bf16* wih_p = (__bf16*)(ws + WS_WIH);
  __bf16* wfc_p = (__bf16*)(ws + WS_WFC);
  float*  bias  = (float*)(ws + WS_BIAS);

  // pack weights into WMMA B-fragment order (bf16)
  pack_b_kernel<<<(G4 / 16) * (HH / 32), 32, 0, stream>>>(W_hh, whh_p, G4, HH);
  pack_b_kernel<<<(G4 / 16) * (FF / 32), 32, 0, stream>>>(W_ih, wih_p, G4, FF);
  pack_b_kernel<<<(PF / 16) * (HH / 32), 32, 0, stream>>>(W_fc, wfc_p, PF, HH);
  bias_kernel<<<(G4 + 255) / 256, 256, 0, stream>>>(b_ih, b_hh, bias);

  // persistent LSTM: one block per 16-row batch tile
  lstm_persistent<<<256 / BT, 256, 0, stream>>>(x, bias, b_fc,
                                                whh_p, wih_p, wfc_p, out);
}
